// FeatureDispersionLoss_24060406792709
// MI455X (gfx1250) — compile-verified, matrix operand order checked by reading, not run on hardware
//
#include <hip/hip_runtime.h>
#include <hip/hip_bf16.h>

typedef __attribute__((ext_vector_type(16))) _Float16 v16h;
typedef __attribute__((ext_vector_type(8)))  _Float16 half8;
typedef __attribute__((ext_vector_type(8)))  float    v8f;

#define FD_D 128           // feature dim (K of the GEMM)
#define FD_BT 128          // block tile: 128x128 output per workgroup
#define FD_MARGIN 0.5f

// ---------------- kernel 0: zero the partial-sum accumulator ----------------
__global__ void fd_zero_kernel(float* acc) {
    if (threadIdx.x == 0 && blockIdx.x == 0) acc[0] = 0.0f;
}

// ---------------- kernel 1: row-normalize f32 -> f16 ------------------------
// One wave (32 lanes) per row of 128 floats; each lane handles 4 elements.
__global__ __launch_bounds__(256)
void fd_normalize_kernel(const float* __restrict__ X, _Float16* __restrict__ Xn, int N) {
    int gid  = blockIdx.x * blockDim.x + threadIdx.x;
    int row  = gid >> 5;        // wave id == row (uniform per wave)
    int lane = gid & 31;
    if (row >= N) return;
    const float4 x4 = *(const float4*)(X + (size_t)row * FD_D + lane * 4);
    float ss = x4.x * x4.x + x4.y * x4.y + x4.z * x4.z + x4.w * x4.w;
    #pragma unroll
    for (int off = 16; off > 0; off >>= 1) ss += __shfl_xor(ss, off, 32);
    float inv = rsqrtf(fmaxf(ss, 1e-16f));
    union { _Float16 h[4]; float2 f2; } u;
    u.h[0] = (_Float16)(x4.x * inv);
    u.h[1] = (_Float16)(x4.y * inv);
    u.h[2] = (_Float16)(x4.z * inv);
    u.h[3] = (_Float16)(x4.w * inv);
    *(float2*)(Xn + (size_t)row * FD_D + lane * 4) = u.f2;
}

// XOR-swizzled LDS panel addressing: row stride = 128 halves (256B), the
// 16-byte chunk index is XORed with (row & 15). 64KB total for both panels,
// conflict behavior equal to a padded layout (b128x32 lanes is 2 bank-cycles
// minimum regardless).
__device__ __forceinline__ const half8* fd_frag(const _Float16* p, int row, int chunk) {
    return (const half8*)(p + row * FD_D + ((chunk ^ (row & 15)) << 3));
}

// ---------------- kernel 2: 128x128 WMMA block GEMM + fused reduction -------
// blockIdx.y = bi (row block), blockIdx.x = bj (col block); only bj >= bi runs.
// 8 waves in a 4x2 grid; each wave owns a 2x4 quad of 16x16 tiles
// (32 rows x 64 cols) -> 1.5 ds_load_b128 per v_wmma.
__global__ __launch_bounds__(256)
void fd_gemm_kernel(const _Float16* __restrict__ Xn, float* __restrict__ acc) {
    const int bi = blockIdx.y, bj = blockIdx.x;
    if (bj < bi) return;   // symmetry: upper block triangle only (uniform exit)

    __shared__ __attribute__((aligned(16))) unsigned char smem[2 * FD_BT * FD_D * 2]; // 64 KB
    _Float16* As = (_Float16*)smem;
    _Float16* Bs = (_Float16*)(smem + FD_BT * FD_D * 2);

    const int tid = threadIdx.x;
    const _Float16* gA = Xn + (size_t)(bi * FD_BT) * FD_D;
    const _Float16* gB = Xn + (size_t)(bj * FD_BT) * FD_D;

    // Stage both 128x128 f16 panels (32 KB each) via async DMA into LDS:
    // GLOBAL_LOAD_ASYNC_TO_LDS_B128, tracked by ASYNCcnt (no VGPR round-trip).
    #pragma unroll
    for (int i = 0; i < 8; ++i) {
        int c   = tid + i * 256;        // 0..2047 16B chunks per panel
        int row = c >> 4;               // 16 chunks per 128-half row
        int cc  = c & 15;
        unsigned sw   = (unsigned)((cc ^ (row & 15)) << 3);
        unsigned ldsA = (unsigned)(uintptr_t)(As + row * FD_D + sw);
        unsigned ldsB = (unsigned)(uintptr_t)(Bs + row * FD_D + sw);
        unsigned voff = (unsigned)((row * FD_D + (cc << 3)) * 2);   // global byte offset
        asm volatile("global_load_async_to_lds_b128 %0, %1, %2"
                     :: "v"(ldsA), "v"(voff), "s"(gA) : "memory");
        asm volatile("global_load_async_to_lds_b128 %0, %1, %2"
                     :: "v"(ldsB), "v"(voff), "s"(gB) : "memory");
    }
    asm volatile("s_wait_asynccnt 0x0" ::: "memory");
    __syncthreads();

    const int lane = tid & 31, wave = tid >> 5;
    const int wr = wave >> 1;           // wave tile-row quad 0..3 (32 rows each)
    const int wc = wave & 1;            // wave tile-col quad 0..1 (64 cols each)
    const int m    = lane & 15;
    const int hsel = (lane >> 4) & 1;   // ISA 16-bit A/B fragment lane split

    const int ar0 = wr * 32 + m;        // A rows for tile-row 0 of the quad
    const int ar1 = ar0 + 16;           // ... tile-row 1
    int br[4];
    #pragma unroll
    for (int t = 0; t < 4; ++t) br[t] = wc * 64 + t * 16 + m;   // B panel rows

    v8f accv[2][4] = {};
    #pragma unroll
    for (int k = 0; k < 4; ++k) {       // K = 128 = 4 x 32
        const int c0 = 4 * k + hsel;    // first 16B chunk of this lane's K slice
        union { v16h v; half8 h[2]; } a0, a1, bb;
        a0.h[0] = *fd_frag(As, ar0, c0); a0.h[1] = *fd_frag(As, ar0, c0 + 2);
        a1.h[0] = *fd_frag(As, ar1, c0); a1.h[1] = *fd_frag(As, ar1, c0 + 2);
        #pragma unroll
        for (int t = 0; t < 4; ++t) {
            bb.h[0] = *fd_frag(Bs, br[t], c0); bb.h[1] = *fd_frag(Bs, br[t], c0 + 2);
            accv[0][t] = __builtin_amdgcn_wmma_f32_16x16x32_f16(false, a0.v, false, bb.v,
                                                               (short)0, accv[0][t], false, false);
            accv[1][t] = __builtin_amdgcn_wmma_f32_16x16x32_f16(false, a1.v, false, bb.v,
                                                               (short)0, accv[1][t], false, false);
        }
    }

    // Epilogue: relu(sim - margin), skip diagonal, weight 2 for off-diag blocks.
    const float wgt = (bi == bj) ? 1.0f : 2.0f;
    float local = 0.0f;
    #pragma unroll
    for (int r2 = 0; r2 < 2; ++r2) {
        const int gib = bi * FD_BT + (wr * 2 + r2) * 16 + (hsel ? 8 : 0); // M = e + 8*(lane/16)
        #pragma unroll
        for (int t = 0; t < 4; ++t) {
            const int gj = bj * FD_BT + (wc * 4 + t) * 16 + m;            // N = lane%16
            #pragma unroll
            for (int e = 0; e < 8; ++e) {
                float v = accv[r2][t][e];
                if (gib + e != gj) local += fmaxf(v - FD_MARGIN, 0.0f);
            }
        }
    }
    local *= wgt;

    __syncthreads();                      // all panel reads done: reuse LDS
    float* red = (float*)smem;
    red[tid] = local;
    __syncthreads();
    #pragma unroll
    for (int s = 128; s > 0; s >>= 1) {
        if (tid < s) red[tid] += red[tid + s];
        __syncthreads();
    }
    if (tid == 0) atomicAdd(acc, red[0]); // global_atomic_add_f32
}

// ---------------- kernel 3: finalize scalar ---------------------------------
__global__ void fd_finalize_kernel(const float* acc, float* out, float scale) {
    if (threadIdx.x == 0 && blockIdx.x == 0) out[0] = acc[0] * scale;
}

extern "C" void kernel_launch(void* const* d_in, const int* in_sizes, int n_in,
                              void* d_out, int out_size, void* d_ws, size_t ws_size,
                              hipStream_t stream) {
    const float* X = (const float*)d_in[0];
    float* out = (float*)d_out;
    const int total = in_sizes[0];      // B*T*D = 32*512*128
    const int N = total / FD_D;         // 16384 rows

    float*     acc = (float*)d_ws;                       // 1 float partial sum
    _Float16*  Xn  = (_Float16*)((char*)d_ws + 256);     // N*128 f16 (4 MB)

    fd_zero_kernel<<<1, 32, 0, stream>>>(acc);

    const int nthreads = N * 32;
    fd_normalize_kernel<<<(nthreads + 255) / 256, 256, 0, stream>>>(X, Xn, N);

    const int nb = N / FD_BT;           // 128 blocks per dim
    dim3 grid(nb, nb);
    fd_gemm_kernel<<<grid, 256, 0, stream>>>(Xn, acc);

    const float scale = 1.0f / ((float)N * (float)N);
    fd_finalize_kernel<<<1, 32, 0, stream>>>(acc, out, scale);
}
